// TriangleMultiplicativeUpdate_32925219291841
// MI455X (gfx1250) — matrix operand order repaired and checked by reading, not compiled, not run on hardware
//
#include <hip/hip_runtime.h>
#include <hip/hip_bf16.h>
#include <math.h>

typedef __bf16 bf16;
typedef __attribute__((ext_vector_type(16))) __bf16 v16bf;
typedef __attribute__((ext_vector_type(8)))  __bf16 v8bf;
typedef __attribute__((ext_vector_type(4)))  __bf16 v4bf;
typedef __attribute__((ext_vector_type(8)))  float   v8f;

#define NSEQ 512
#define CH   128
#define PL   262144   // NSEQ*NSEQ = rows = per-channel plane size

union ABFrag { v16bf v; v8bf h[2]; };

__device__ __forceinline__ float sigf(float x) { return 1.0f / (1.0f + __expf(-x)); }

// ---------------------------------------------------------------------------
// Kernel 1: LayerNorm(z) -> zn bf16 [r][c], one wave per 128-ch row.
// ---------------------------------------------------------------------------
__global__ void __launch_bounds__(256, 1)
k_ln_in(const float* __restrict__ z, const float* __restrict__ nw,
        const float* __restrict__ nb, bf16* __restrict__ zn) {
  const int lane = threadIdx.x & 31;
  const int wv   = threadIdx.x >> 5;
  const int r    = blockIdx.x * 8 + wv;
  float4 x = *(const float4*)(z + (size_t)r * CH + lane * 4);
  float s = x.x + x.y + x.z + x.w;
  float q = x.x*x.x + x.y*x.y + x.z*x.z + x.w*x.w;
#pragma unroll
  for (int off = 16; off > 0; off >>= 1) {
    s += __shfl_xor(s, off, 32);
    q += __shfl_xor(q, off, 32);
  }
  float mean = s * (1.0f / 128.0f);
  float var  = q * (1.0f / 128.0f) - mean * mean;
  float rstd = rsqrtf(var + 1e-5f);
  float4 w4 = *(const float4*)(nw + lane * 4);
  float4 b4 = *(const float4*)(nb + lane * 4);
  v4bf o;
  o[0] = (bf16)((x.x - mean) * rstd * w4.x + b4.x);
  o[1] = (bf16)((x.y - mean) * rstd * w4.y + b4.y);
  o[2] = (bf16)((x.z - mean) * rstd * w4.z + b4.z);
  o[3] = (bf16)((x.w - mean) * rstd * w4.w + b4.w);
  *(v4bf*)(zn + (size_t)r * CH + lane * 4) = o;
}

// ---------------------------------------------------------------------------
// Kernel 2: repack 6 weight matrices (f32 [n][k], 128x128) into bf16 WMMA
// B-fragment blocked layout: wblk[w][kb][jt][lane][16]
// ---------------------------------------------------------------------------
__global__ void k_repack(const float* w0, const float* w1, const float* w2,
                         const float* w3, const float* w4, const float* w5,
                         bf16* __restrict__ wblk) {
  int idx = blockIdx.x * 256 + threadIdx.x;   // < 6*16384 = 98304
  const float* ws[6] = {w0, w1, w2, w3, w4, w5};
  int w    = idx >> 14;
  int rem  = idx & 16383;
  int s    = rem & 15;
  int lane = (rem >> 4) & 31;
  int jt   = (rem >> 9) & 7;
  int kb   = (rem >> 12) & 3;
  int n = jt * 16 + (lane & 15);
  int k = kb * 32 + (lane >> 4) * 8 + (s & 7) + (s >> 3) * 16;
  wblk[idx] = (bf16)ws[w][n * CH + k];
}

// ---------------------------------------------------------------------------
// 4-tile GEMM helper, double-buffered B fragments:
//   loads for kb+1 issue while kb's WMMAs execute (one wait per K-step).
// ---------------------------------------------------------------------------
__device__ __forceinline__ void gemm4(const bf16* __restrict__ wblk, int w, int jh,
                                      const ABFrag A[4], v8f acc[4], int lane) {
  const bf16* base = wblk + ((w * 4) * 8 + jh * 4) * 512 + lane * 16;
  ABFrag B[2][4];
#pragma unroll
  for (int q = 0; q < 4; ++q) {
    acc[q] = (v8f){0.f,0.f,0.f,0.f,0.f,0.f,0.f,0.f};
    B[0][q].h[0] = *(const v8bf*)(base + q * 512);
    B[0][q].h[1] = *(const v8bf*)(base + q * 512 + 8);
  }
#pragma unroll
  for (int kb = 0; kb < 4; ++kb) {
    if (kb < 3) {
#pragma unroll
      for (int q = 0; q < 4; ++q) {
        B[(kb + 1) & 1][q].h[0] = *(const v8bf*)(base + (kb + 1) * 4096 + q * 512);
        B[(kb + 1) & 1][q].h[1] = *(const v8bf*)(base + (kb + 1) * 4096 + q * 512 + 8);
      }
    }
#pragma unroll
    for (int q = 0; q < 4; ++q)
      acc[q] = __builtin_amdgcn_wmma_f32_16x16x32_bf16(
          false, A[kb].v, false, B[kb & 1][q].v, (short)0, acc[q], false, false);
  }
}

// ---------------------------------------------------------------------------
// Kernel 3: fused projections. One wave per 16-row tile (rows r = i*512+k).
// ---------------------------------------------------------------------------
__global__ void __launch_bounds__(256, 1)
k_proj(const bf16* __restrict__ zn, const float* __restrict__ mask,
       const bf16* __restrict__ wblk, bf16* __restrict__ a_t,
       bf16* __restrict__ b_blk, bf16* __restrict__ g_t) {
  const int lane = threadIdx.x & 31;
  const int wv   = threadIdx.x >> 5;
  const int tile = blockIdx.x * 8 + wv;
  const int r0   = tile * 16;
  const int i    = r0 >> 9;           // = i for 'a' rows, = j for 'b' rows
  const int k0   = r0 & 511;
  const int mrow = lane & 15;
  const int kh   = lane >> 4;
  const int s0   = kh * 8;

  ABFrag A[4];
#pragma unroll
  for (int kb = 0; kb < 4; ++kb) {
    const bf16* ap = zn + (r0 + mrow) * CH + kb * 32 + kh * 8;
    A[kb].h[0] = *(const v8bf*)ap;
    A[kb].h[1] = *(const v8bf*)(ap + 16);
  }

  float4 mk0 = *(const float4*)(mask + r0 + s0);
  float4 mk1 = *(const float4*)(mask + r0 + s0 + 4);
  float mk[8] = {mk0.x, mk0.y, mk0.z, mk0.w, mk1.x, mk1.y, mk1.z, mk1.w};

  const int kb_s   = k0 >> 5;
  const int khalf  = (k0 >> 4) & 1;
  const int jt_b   = i >> 4;
  const int lane_b = khalf * 16 + (i & 15);

#pragma unroll 1
  for (int jh = 0; jh < 2; ++jh) {
    v8f accp[4], accg[4];

    // a = mask * (zn@apW^T) * sigmoid(zn@agW^T)  -> a_t[c][i][k]
    gemm4(wblk, 0, jh, A, accp, lane);
    gemm4(wblk, 1, jh, A, accg, lane);
#pragma unroll
    for (int q = 0; q < 4; ++q) {
      int c = (jh * 4 + q) * 16 + (lane & 15);
      v8bf o;
#pragma unroll
      for (int v = 0; v < 8; ++v)
        o[v] = (bf16)(mk[v] * accp[q][v] * sigf(accg[q][v]));
      *(v8bf*)(a_t + c * PL + i * 512 + k0 + s0) = o;
    }

    // b = mask * (zn@bpW^T) * sigmoid(zn@bgW^T) -> blocked B-frag layout
    gemm4(wblk, 2, jh, A, accp, lane);
    gemm4(wblk, 3, jh, A, accg, lane);
#pragma unroll
    for (int q = 0; q < 4; ++q) {
      int c = (jh * 4 + q) * 16 + (lane & 15);
      v8bf o;
#pragma unroll
      for (int v = 0; v < 8; ++v)
        o[v] = (bf16)(mk[v] * accp[q][v] * sigf(accg[q][v]));
      *(v8bf*)(b_blk + ((c * 16 + kb_s) * 32 + jt_b) * 512 + lane_b * 16 + s0) = o;
    }

    // g = sigmoid(zn@gW^T) -> g_t[c][r]
    gemm4(wblk, 4, jh, A, accp, lane);
#pragma unroll
    for (int q = 0; q < 4; ++q) {
      int c = (jh * 4 + q) * 16 + (lane & 15);
      v8bf o;
#pragma unroll
      for (int v = 0; v < 8; ++v)
        o[v] = (bf16)sigf(accp[q][v]);
      *(v8bf*)(g_t + c * PL + r0 + s0) = o;
    }
  }
}

// ---------------------------------------------------------------------------
// Kernel 4: triangle einsum. Per channel c: P_c = A_c x B_c^T.
// Block = 8 waves = 128x128 output tile. B panels streamed into LDS with
// double-buffered GLOBAL_LOAD_ASYNC_TO_LDS_B128 (ASYNCcnt), overlapping WMMA.
// ---------------------------------------------------------------------------
__global__ void __launch_bounds__(256, 1)
k_tri(const bf16* __restrict__ a_t, const bf16* __restrict__ b_blk,
      float* __restrict__ p) {
  const int tid  = threadIdx.x;
  const int lane = tid & 31;
  const int wv   = tid >> 5;
  const int jt0  = blockIdx.x;     // 128-wide j block
  const int it0  = blockIdx.y;     // 128-wide i block
  const int c    = blockIdx.z;
  const int i0   = it0 * 128 + wv * 16;
  const int j0   = jt0 * 128;
  const int mrow = lane & 15;
  const int kh   = lane >> 4;

  __shared__ __align__(16) bf16 smB[2][4096];   // 2 x (8 B-frag chunks, 8KB)
  v8f acc[8] = {};

  const unsigned lds0 = (unsigned)(uintptr_t)(&smB[0][0]);
  const unsigned vo0  = (unsigned)(tid * 16);
  const bf16* abase = a_t + c * PL + (i0 + mrow) * 512 + kh * 16;
  const bf16* bbase = b_blk + c * PL + jt0 * 4096;   // + kb*16384 elements

  // async copy of one 8KB B panel (8 jt chunks) into smB[buf]
  auto issue = [&](int buf, int kb) {
    const bf16* src = bbase + kb * 16384;
    unsigned lds = lds0 + (unsigned)(buf * 8192) + vo0;
    asm volatile("global_load_async_to_lds_b128 %0, %1, %2"
                 :: "v"(lds), "v"(vo0), "s"(src) : "memory");
    asm volatile("global_load_async_to_lds_b128 %0, %1, %2 offset:4096"
                 :: "v"(lds), "v"(vo0), "s"(src) : "memory");
  };

  issue(0, 0);
#pragma unroll 1
  for (int kb = 0; kb < 16; ++kb) {
    if (kb < 15) {
      issue((kb + 1) & 1, kb + 1);
      asm volatile("s_wait_asynccnt 0x2" ::: "memory");  // panel kb landed
    } else {
      asm volatile("s_wait_asynccnt 0x0" ::: "memory");
    }
    __syncthreads();

    ABFrag Af;
    const bf16* ap = abase + kb * 32;
    Af.h[0] = *(const v8bf*)ap;
    Af.h[1] = *(const v8bf*)(ap + 8);

    const bf16* sb = &smB[kb & 1][0] + lane * 16;
    ABFrag Bf[8];
#pragma unroll
    for (int q = 0; q < 8; ++q) {
      Bf[q].h[0] = *(const v8bf*)(sb + q * 512);
      Bf[q].h[1] = *(const v8bf*)(sb + q * 512 + 8);
    }
#pragma unroll
    for (int q = 0; q < 8; ++q)
      acc[q] = __builtin_amdgcn_wmma_f32_16x16x32_bf16(
          false, Af.v, false, Bf[q].v, (short)0, acc[q], false, false);
    __syncthreads();
  }

  const int s0r = kh * 8;
#pragma unroll
  for (int q = 0; q < 8; ++q) {
    int jcol = j0 + q * 16 + (lane & 15);
#pragma unroll
    for (int v = 0; v < 8; ++v)
      p[c * PL + (i0 + s0r + v) * 512 + jcol] = acc[q][v];
  }
}

// ---------------------------------------------------------------------------
// Kernel 5: LayerNorm over c of p[c][i][j] -> pn bf16 [r][c] via LDS transpose.
// ---------------------------------------------------------------------------
__global__ void __launch_bounds__(128, 1)
k_ln_out(const float* __restrict__ p, const float* __restrict__ nw,
         const float* __restrict__ nb, bf16* __restrict__ pn) {
  __shared__ __align__(16) bf16 smT[128 * 136];
  const int tid = threadIdx.x;
  const int r   = blockIdx.x * 128 + tid;
  float s = 0.f, sq = 0.f;
  for (int c2 = 0; c2 < 128; ++c2) {
    float v = p[c2 * PL + r];
    s += v; sq += v * v;
  }
  float mean = s * (1.0f / 128.0f);
  float var  = sq * (1.0f / 128.0f) - mean * mean;
  float rstd = rsqrtf(var + 1e-5f);
  for (int c2 = 0; c2 < 128; ++c2) {
    float v = p[c2 * PL + r];
    smT[tid * 136 + c2] = (bf16)((v - mean) * rstd * nw[c2] + nb[c2]);
  }
  __syncthreads();
  const int rbase = blockIdx.x * 128;
#pragma unroll
  for (int pass = 0; pass < 16; ++pass) {
    int rr = pass * 8 + (tid >> 4);
    int cp = tid & 15;
    *(v8bf*)(pn + (rbase + rr) * CH + cp * 8) = *(const v8bf*)(smT + rr * 136 + cp * 8);
  }
}

// ---------------------------------------------------------------------------
// Kernel 6: out = (pn @ z_w^T) * g   (gate pre-stored as sigmoid in g_t[c][r])
// ---------------------------------------------------------------------------
__global__ void __launch_bounds__(256, 1)
k_out(const bf16* __restrict__ pn, const bf16* __restrict__ wblk,
      const bf16* __restrict__ g_t, float* __restrict__ out) {
  const int lane = threadIdx.x & 31;
  const int wv   = threadIdx.x >> 5;
  const int tile = blockIdx.x * 8 + wv;
  const int r0   = tile * 16;
  const int mrow = lane & 15;
  const int kh   = lane >> 4;
  const int s0   = kh * 8;

  ABFrag A[4];
#pragma unroll
  for (int kb = 0; kb < 4; ++kb) {
    const bf16* ap = pn + (r0 + mrow) * CH + kb * 32 + kh * 8;
    A[kb].h[0] = *(const v8bf*)ap;
    A[kb].h[1] = *(const v8bf*)(ap + 16);
  }

#pragma unroll 1
  for (int jh = 0; jh < 2; ++jh) {
    v8f acc[4];
    gemm4(wblk, 5, jh, A, acc, lane);
#pragma unroll
    for (int q = 0; q < 4; ++q) {
      int c = (jh * 4 + q) * 16 + (lane & 15);
      v8bf g8 = *(const v8bf*)(g_t + c * PL + r0 + s0);
#pragma unroll
      for (int v = 0; v < 8; ++v)
        out[(r0 + s0 + v) * CH + c] = acc[q][v] * (float)g8[v];
    }
  }
}

// ---------------------------------------------------------------------------
extern "C" void kernel_launch(void* const* d_in, const int* in_sizes, int n_in,
                              void* d_out, int out_size, void* d_ws, size_t ws_size,
                              hipStream_t stream) {
  const float* z    = (const float*)d_in[0];
  const float* mask = (const float*)d_in[1];
  const float* niw  = (const float*)d_in[2];
  const float* nib  = (const float*)d_in[3];
  const float* w_ap = (const float*)d_in[4];
  const float* w_ag = (const float*)d_in[5];
  const float* w_bp = (const float*)d_in[6];
  const float* w_bg = (const float*)d_in[7];
  const float* now_ = (const float*)d_in[8];
  const float* nob  = (const float*)d_in[9];
  const float* w_z  = (const float*)d_in[10];
  const float* w_g  = (const float*)d_in[11];
  float* out = (float*)d_out;

  char* ws = (char*)d_ws;
  bf16*  zn    = (bf16*)(ws);                          // 64 MB (reused as pn)
  bf16*  a_t   = (bf16*)(ws + (size_t)67108864);       // 64 MB
  bf16*  b_blk = (bf16*)(ws + (size_t)134217728);      // 64 MB
  bf16*  g_t   = (bf16*)(ws + (size_t)201326592);      // 64 MB
  float* p     = (float*)(ws + (size_t)268435456);     // 128 MB
  bf16*  wblk  = (bf16*)(ws + (size_t)402653184);      // 192 KB

  k_ln_in <<<PL / 8, 256, 0, stream>>>(z, niw, nib, zn);
  k_repack<<<384, 256, 0, stream>>>(w_ap, w_ag, w_bp, w_bg, w_g, w_z, wblk);
  k_proj  <<<PL / 128, 256, 0, stream>>>(zn, mask, wblk, a_t, b_blk, g_t);
  k_tri   <<<dim3(4, 4, 128), 256, 0, stream>>>(a_t, b_blk, p);
  k_ln_out<<<PL / 128, 128, 0, stream>>>(p, now_, nob, zn /* pn */);
  k_out   <<<PL / 128, 256, 0, stream>>>(zn /* pn */, wblk, g_t, out);
}
